// CrossAttention_32263794328065
// MI455X (gfx1250) — matrix operand order repaired and checked by reading, not compiled, not run on hardware
//
#include <hip/hip_runtime.h>

// ---------------------------------------------------------------------------
// Types for CDNA5 WMMA (gfx1250, wave32)
// ---------------------------------------------------------------------------
typedef __attribute__((ext_vector_type(16))) __bf16 bf16x16;
typedef __attribute__((ext_vector_type(8)))  float  f32x8;

struct U16x16 { uint4 lo, hi; };   // 32B, bit-castable to bf16x16

__device__ __forceinline__ unsigned short f2bu(float f) {
    unsigned u = __builtin_bit_cast(unsigned, f);
    u += 0x7fffu + ((u >> 16) & 1u);          // round-to-nearest-even
    return (unsigned short)(u >> 16);
}
__device__ __forceinline__ float bu2f(unsigned short h) {
    unsigned u = ((unsigned)h) << 16;
    return __builtin_bit_cast(float, u);
}
__device__ __forceinline__ __bf16 us2b(unsigned short h) {
    return __builtin_bit_cast(__bf16, h);
}

__device__ __forceinline__ f32x8 wmma_bf16(bf16x16 a, bf16x16 b, f32x8 c) {
    // D = A(16x32) * B(32x16) + C(16x16), f32 accum
    return __builtin_amdgcn_wmma_f32_16x16x32_bf16(
        false, a, false, b, (short)0, c, false, false);
}

// LDS byte offset of a __shared__ object (generic -> addrspace(3) -> int)
__device__ __forceinline__ unsigned lds_off(const void* p) {
    return (unsigned)(unsigned long long)(
        (__attribute__((address_space(3))) const char*)p);
}

// Async 16B copy global -> LDS (per-lane addresses), tracked by ASYNCcnt.
__device__ __forceinline__ void async_ld_b128(unsigned lds_byte_off,
                                              const void* gaddr) {
    unsigned long long ga = (unsigned long long)gaddr;
    asm volatile("global_load_async_to_lds_b128 %0, %1, off"
                 :: "v"(lds_byte_off), "v"(ga)
                 : "memory");
}
__device__ __forceinline__ void wait_asynccnt0() {
    asm volatile("s_wait_asynccnt 0" ::: "memory");
}

// ---------------------------------------------------------------------------
// Elementwise f32 -> bf16 conversion (n multiple of 4)
// ---------------------------------------------------------------------------
__global__ void cvt_f32_bf16_k(const float* __restrict__ in,
                               unsigned short* __restrict__ out, int n) {
    int i4 = (blockIdx.x * blockDim.x + threadIdx.x) * 4;
    if (i4 >= n) return;
    float4 v = *(const float4*)&in[i4];
    uint2 p;
    p.x = (unsigned)f2bu(v.x) | ((unsigned)f2bu(v.y) << 16);
    p.y = (unsigned)f2bu(v.z) | ((unsigned)f2bu(v.w) << 16);
    *(uint2*)&out[i4] = p;
}

// ---------------------------------------------------------------------------
// RoPE in place on bf16 tensor x[B,S,NH*64].
// Faithful to reference: "cos_pos" holds sin(pe), "sin_pos" holds cos(pe).
// ---------------------------------------------------------------------------
__global__ void rope_inplace_k(unsigned short* __restrict__ x,
                               int S, int NH, int total_pairs) {
    int p = blockIdx.x * blockDim.x + threadIdx.x;
    if (p >= total_pairs) return;
    int j    = p & 31;           // 0..31 (head_dim/2 = 32)
    int rest = p >> 5;
    int h    = rest % NH;
    int s    = (rest / NH) % S;
    int b    = rest / (NH * S);
    size_t base = (((size_t)b * S + s) * NH + h) * 64 + 2 * j;
    float x0 = bu2f(x[base]), x1 = bu2f(x[base + 1]);
    float theta = __powf(10000.0f, -(float)(2 * j) * (1.0f / 64.0f));
    float pe = (float)s * theta;
    float sp = __sinf(pe), cp = __cosf(pe);
    x[base]     = f2bu(x0 * sp - x1 * cp);
    x[base + 1] = f2bu(x1 * sp + x0 * cp);
}

// ---------------------------------------------------------------------------
// Tiled WMMA GEMM: C[M,N] = A[M,K](bf16) @ B[K,N](bf16) + bias[N]
// Block tile 128x128, K-step 32, 256 threads = 8 waves (2 M-groups x 4 N-groups)
// A staged via async global->LDS; B staged transposed for b128 fragments.
// ---------------------------------------------------------------------------
__device__ __forceinline__ void storeC(float* p, float v) { *p = v; }
__device__ __forceinline__ void storeC(unsigned short* p, float v) { *p = f2bu(v); }

template <typename OutT>
__global__ __launch_bounds__(256) void gemm_bf16_k(
    const unsigned short* __restrict__ A,  // [M,K]
    const unsigned short* __restrict__ B,  // [K,N]
    const float* __restrict__ bias,        // [N]
    OutT* __restrict__ C,                  // [M,N]
    int M, int N, int K) {
    constexpr int ALD = 40;                // padded strides (20 dwords: no bank conflicts)
    constexpr int BLD = 40;
    __shared__ __align__(16) unsigned short As[128 * ALD];   // [m][k]
    __shared__ __align__(16) unsigned short Bst[128 * BLD];  // [n][k] (transposed)

    const int tid  = threadIdx.x;
    const int wave = tid >> 5, lane = tid & 31;
    const int half = lane >> 4, ln = lane & 15;
    const int m0 = blockIdx.y * 128, n0 = blockIdx.x * 128;
    const int wm = wave >> 2;              // 0..1 -> 64 rows
    const int wn = wave & 3;               // 0..3 -> 32 cols

    f32x8 acc[4][2];
#pragma unroll
    for (int i = 0; i < 4; i++)
#pragma unroll
        for (int j = 0; j < 2; j++)
#pragma unroll
            for (int r = 0; r < 8; r++) acc[i][j][r] = 0.0f;

    for (int kk = 0; kk < K; kk += 32) {
        __syncthreads();
        // stage A tile 128x32 via async global->LDS (per-lane 16B chunks)
        for (int q = tid; q < 512; q += 256) {
            int r = q >> 2, c = (q & 3) * 8;
            async_ld_b128(lds_off(&As[r * ALD + c]),
                          &A[(size_t)(m0 + r) * K + kk + c]);
            if (kk + 32 < K)
                __builtin_prefetch(&A[(size_t)(m0 + r) * K + kk + 32 + c], 0, 0);
        }
        // stage B tile 32x128 transposed -> Bst[n][k]
        for (int q = tid; q < 512; q += 256) {
            int r = q >> 4, c = (q & 15) * 8;   // r = k-row, c = n-col chunk
            int gk = kk + r, gn = n0 + c;
            if (gn + 7 < N) {
                uint4 v = *(const uint4*)&B[(size_t)gk * N + gn];
                unsigned short e[8];
                *(uint4*)e = v;
#pragma unroll
                for (int t = 0; t < 8; t++) Bst[(c + t) * BLD + r] = e[t];
                if (kk + 32 < K)
                    __builtin_prefetch(&B[(size_t)(gk + 32) * N + gn], 0, 0);
            } else {
#pragma unroll
                for (int t = 0; t < 8; t++)
                    Bst[(c + t) * BLD + r] =
                        (gn + t < N) ? B[(size_t)gk * N + gn + t] : (unsigned short)0;
            }
        }
        wait_asynccnt0();
        __syncthreads();

        // B fragments: contiguous 16 bf16 along k  -> 2x ds_load_b128
        bf16x16 bfr[2];
#pragma unroll
        for (int ns = 0; ns < 2; ns++)
#pragma unroll
            for (int i = 0; i < 16; i++) {
                int k = i + 16 * half;     // contraction index 0..31
                bfr[ns][i] = us2b(Bst[(wn * 32 + ns * 16 + ln) * BLD + k]);
            }
#pragma unroll
        for (int ms = 0; ms < 4; ms++) {
            bf16x16 af;
#pragma unroll
            for (int i = 0; i < 16; i++) {
                int k = i + 16 * half;
                af[i] = us2b(As[(wm * 64 + ms * 16 + ln) * ALD + k]);
            }
            acc[ms][0] = wmma_bf16(af, bfr[0], acc[ms][0]);
            acc[ms][1] = wmma_bf16(af, bfr[1], acc[ms][1]);
        }
    }

#pragma unroll
    for (int ms = 0; ms < 4; ms++)
#pragma unroll
        for (int ns = 0; ns < 2; ns++)
#pragma unroll
            for (int r = 0; r < 8; r++) {
                int row = m0 + wm * 64 + ms * 16 + r + 8 * half;
                int col = n0 + wn * 32 + ns * 16 + ln;
                if (col < N)
                    storeC(&C[(size_t)row * N + col], acc[ms][ns][r] + bias[col]);
            }
}

// ---------------------------------------------------------------------------
// Flash attention (MLA): one (b,h), 128 query rows per block, 64-key tiles.
// 8 waves x 16 rows each; online softmax wave-local (one pass per 64 keys).
// Q fragments register-resident (loaded from global). K double-buffered via
// async global->LDS; V single-buffered (next tile rides in registers during
// compute, scattered after the post-PV barrier). scale = 16.
// ---------------------------------------------------------------------------
__global__ __launch_bounds__(256) void mla_attn_k(
    const unsigned short* __restrict__ qtc,  // [B,S,1024] q nope
    const unsigned short* __restrict__ qr,   // [B,S,1024] q rope (rotated)
    const unsigned short* __restrict__ ktc,  // [B,S,1024] k nope
    const unsigned short* __restrict__ kr,   // [B,S,64]   k rope (rotated, shared)
    const unsigned short* __restrict__ vtc,  // [B,S,1024] v
    unsigned short* __restrict__ out,        // [B,S,1024]
    int S) {
    constexpr int KLD = 136;                 // 272B rows: 16B aligned, conflict-free
    constexpr int VLD = 72;                  // Vt[dim][key] padded (144B rows)
    constexpr int PLD = 72;                  // Ps padded: aligned b128 A-frag loads
    __shared__ __align__(16) unsigned short Ks[2][64 * KLD];   // [key][dim]
    __shared__ __align__(16) unsigned short Vt[64 * VLD];      // [dim][key]
    __shared__ __align__(16) unsigned short Ps[8 * 16 * PLD];

    const int b = blockIdx.z, h = blockIdx.y;
    const int q0 = blockIdx.x * 128;
    const int tid  = threadIdx.x;
    const int wave = tid >> 5, lane = tid & 31;
    const int half = lane >> 4, ln = lane & 15;

    // ---- Q A-fragments straight from global (two 16B runs per 32-K chunk) ----
    bf16x16 qa[4];
    {
        size_t qrow = ((size_t)b * S + (q0 + wave * 16 + ln)) * 1024 + (size_t)h * 64;
#pragma unroll
        for (int kb = 0; kb < 4; kb++) {
            int base = kb * 32 + 8 * half;
            const unsigned short* src =
                (kb < 2) ? &qtc[qrow + base] : &qr[qrow + base - 64];
            U16x16 t;
            t.lo = *(const uint4*)src;          // elements i=0..7  (k=base..base+7)
            t.hi = *(const uint4*)(src + 16);   // elements i=8..15 (k=base+16..+23)
            qa[kb] = __builtin_bit_cast(bf16x16, t);
        }
    }

    // ---- online-softmax state (8 rows per lane: row = r + 8*half) ----
    float mrow[8], lrow[8];
    f32x8 o[4];
#pragma unroll
    for (int r = 0; r < 8; r++) { mrow[r] = -1e30f; lrow[r] = 0.0f; }
#pragma unroll
    for (int t = 0; t < 4; t++)
#pragma unroll
        for (int r = 0; r < 8; r++) o[t][r] = 0.0f;

    const float inv_scale = 1.0f / 16.0f;    // 1/(sqrt(64)+sqrt(64))
    const int nT = S / 64;

    uint2 vreg[4];
    // K tile: 64 keys x 128 dims = 1024 16B-chunks -> 4 async per thread
    auto stage_K = [&](int tile, int bufn) {
#pragma unroll
        for (int u = 0; u < 4; u++) {
            int q = tid + u * 256;
            int n = q >> 4;                  // key 0..63
            int c = (q & 15) * 8;            // dim chunk 0..120
            size_t srow = (size_t)b * S + (tile * 64 + n);
            const void* src = (c < 64)
                ? (const void*)&ktc[srow * 1024 + (size_t)h * 64 + c]
                : (const void*)&kr[srow * 64 + (c - 64)];
            async_ld_b128(lds_off(&Ks[bufn][n * KLD + c]), src);
        }
    };
    // V tile: 64 keys x 64 dims = 1024 8B-chunks -> 4 regs per thread
    auto load_V = [&](int tile) {
#pragma unroll
        for (int u = 0; u < 4; u++) {
            int q = tid + u * 256;
            int n = q >> 4, c4 = (q & 15) * 4;
            vreg[u] = *(const uint2*)&vtc[((size_t)b * S + (tile * 64 + n)) * 1024 +
                                          (size_t)h * 64 + c4];
        }
    };
    auto scatter_V = [&]() {
#pragma unroll
        for (int u = 0; u < 4; u++) {
            int q = tid + u * 256;
            int n = q >> 4, c4 = (q & 15) * 4;
            unsigned short e[4];
            *(uint2*)e = vreg[u];
#pragma unroll
            for (int t = 0; t < 4; t++) Vt[(c4 + t) * VLD + n] = e[t];
        }
    };

    // ---- prologue: stage tile 0 ----
    stage_K(0, 0);
    load_V(0);
    scatter_V();
    wait_asynccnt0();
    __syncthreads();

    for (int it = 0; it < nT; ++it) {
        const int bufn = it & 1;
        // kick off next tile (overlaps with this tile's compute)
        if (it + 1 < nT) {
            stage_K(it + 1, bufn ^ 1);
            load_V(it + 1);
            if (it + 2 < nT) {
                int q = tid, n = q >> 4, c = (q & 15) * 8;
                size_t srow = (size_t)b * S + ((it + 2) * 64 + n);
                __builtin_prefetch(&ktc[srow * 1024 + (size_t)h * 64 + c], 0, 0);
                __builtin_prefetch(&vtc[srow * 1024 + (size_t)h * 64 + c], 0, 0);
            }
        }

        // ---- scores: four 16x16 tiles over 64 keys ----
        f32x8 sc[4];
#pragma unroll
        for (int nt = 0; nt < 4; nt++) {
            f32x8 c;
#pragma unroll
            for (int r = 0; r < 8; r++) c[r] = 0.0f;
#pragma unroll
            for (int kb = 0; kb < 4; kb++) {
                bf16x16 kf;
#pragma unroll
                for (int i = 0; i < 16; i++) {
                    int k = kb * 32 + i + 8 * half + ((i >= 8) ? 8 : 0); // contraction dim
                    kf[i] = us2b(Ks[bufn][(nt * 16 + ln) * KLD + k]); // key = B column
                }
                c = wmma_bf16(qa[kb], kf, c);
            }
#pragma unroll
            for (int r = 0; r < 8; r++) c[r] *= inv_scale;
            sc[nt] = c;
        }

        // ---- row max over 64 keys (4 frags + 16-lane reduce) ----
        float fac[8];
#pragma unroll
        for (int r = 0; r < 8; r++) {
            float v = fmaxf(fmaxf(sc[0][r], sc[1][r]), fmaxf(sc[2][r], sc[3][r]));
            v = fmaxf(v, __shfl_xor(v, 1, 16));
            v = fmaxf(v, __shfl_xor(v, 2, 16));
            v = fmaxf(v, __shfl_xor(v, 4, 16));
            v = fmaxf(v, __shfl_xor(v, 8, 16));
            float mn = fmaxf(mrow[r], v);
            fac[r] = __expf(mrow[r] - mn);
            mrow[r] = mn;
        }
#pragma unroll
        for (int t = 0; t < 4; t++)
#pragma unroll
            for (int r = 0; r < 8; r++) o[t][r] *= fac[r];

        // ---- P = exp(s - m), row sums, store P to LDS as bf16 ----
#pragma unroll
        for (int nt = 0; nt < 4; nt++)
#pragma unroll
            for (int r = 0; r < 8; r++)
                sc[nt][r] = __expf(sc[nt][r] - mrow[r]);
#pragma unroll
        for (int r = 0; r < 8; r++) {
            float v = (sc[0][r] + sc[1][r]) + (sc[2][r] + sc[3][r]);
            v += __shfl_xor(v, 1, 16);
            v += __shfl_xor(v, 2, 16);
            v += __shfl_xor(v, 4, 16);
            v += __shfl_xor(v, 8, 16);
            lrow[r] = lrow[r] * fac[r] + v;
        }
#pragma unroll
        for (int nt = 0; nt < 4; nt++)
#pragma unroll
            for (int r = 0; r < 8; r++)
                Ps[wave * 16 * PLD + (r + 8 * half) * PLD + nt * 16 + ln] =
                    f2bu(sc[nt][r]);
        // wave-local LDS: DS ops from one wave complete in order -> no barrier

        // ---- O += P @ V : P 16x64 = two A-frags; V frags from Vt as two 8-key runs ----
        bf16x16 pa[2];
#pragma unroll
        for (int kc2 = 0; kc2 < 2; kc2++)
#pragma unroll
            for (int i = 0; i < 16; i++) {
                int k = kc2 * 32 + i + 8 * half + ((i >= 8) ? 8 : 0);
                pa[kc2][i] = us2b(Ps[wave * 16 * PLD + ln * PLD + k]);
            }
#pragma unroll
        for (int nt2 = 0; nt2 < 4; nt2++) {
#pragma unroll
            for (int kc2 = 0; kc2 < 2; kc2++) {
                bf16x16 vb;
#pragma unroll
                for (int i = 0; i < 16; i++) {
                    int k = kc2 * 32 + i + 8 * half + ((i >= 8) ? 8 : 0);      // key index 0..63
                    vb[i] = us2b(Vt[(nt2 * 16 + ln) * VLD + k]);
                }
                o[nt2] = wmma_bf16(pa[kc2], vb, o[nt2]);
            }
        }

        // all waves finished reading Vt -> safe to overwrite with next tile
        __syncthreads();
        if (it + 1 < nT) scatter_V();
        wait_asynccnt0();
        __syncthreads();
    }

    // ---- normalize and store (out layout [B,S,H*64]) ----
    float invl[8];
#pragma unroll
    for (int r = 0; r < 8; r++) invl[r] = 1.0f / lrow[r];
#pragma unroll
    for (int nt2 = 0; nt2 < 4; nt2++)
#pragma unroll
        for (int r = 0; r < 8; r++) {
            int row = q0 + wave * 16 + r + 8 * half;
            int col = h * 64 + nt2 * 16 + ln;
            out[((size_t)b * S + row) * 1024 + col] = f2bu(o[nt2][r] * invl[r]);
        }
}

// ---------------------------------------------------------------------------
// Host orchestration
// ---------------------------------------------------------------------------
extern "C" void kernel_launch(void* const* d_in, const int* in_sizes, int n_in,
                              void* d_out, int out_size, void* d_ws, size_t ws_size,
                              hipStream_t stream) {
    (void)in_sizes; (void)n_in; (void)out_size; (void)ws_size;
    const int B = 2, S = 2048, Z = 1024, D = 512, U = 1024, H = 16;
    const int MS = B * S;  // 4096 token rows

    const float* query = (const float*)d_in[0];
    const float* key   = (const float*)d_in[1];
    // d_in[2] (value) unused by the reference
    const float* w_dkv = (const float*)d_in[3];
    const float* b_dkv = (const float*)d_in[4];
    const float* w_uk  = (const float*)d_in[5];
    const float* b_uk  = (const float*)d_in[6];
    const float* w_uv  = (const float*)d_in[7];
    const float* b_uv  = (const float*)d_in[8];
    const float* w_dq  = (const float*)d_in[9];
    const float* b_dq  = (const float*)d_in[10];
    const float* w_uq  = (const float*)d_in[11];
    const float* b_uq  = (const float*)d_in[12];
    const float* w_qr  = (const float*)d_in[13];
    const float* b_qr  = (const float*)d_in[14];
    const float* w_kr  = (const float*)d_in[15];
    const float* b_kr  = (const float*)d_in[16];
    const float* w_fc  = (const float*)d_in[17];
    const float* b_fc  = (const float*)d_in[18];
    float* outp = (float*)d_out;

    // workspace carve-out (bf16 buffers)
    char* ws = (char*)d_ws;
    size_t off = 0;
    auto take = [&](size_t elems) -> unsigned short* {
        unsigned short* p = (unsigned short*)(ws + off);
        off += ((elems * 2 + 255) / 256) * 256;
        return p;
    };
    unsigned short* key_bf   = take((size_t)MS * Z);
    unsigned short* query_bf = take((size_t)MS * Z);
    unsigned short* wdkv_bf  = take((size_t)Z * D);
    unsigned short* wuk_bf   = take((size_t)D * U);
    unsigned short* wuv_bf   = take((size_t)D * U);
    unsigned short* wdq_bf   = take((size_t)Z * D);
    unsigned short* wuq_bf   = take((size_t)D * U);
    unsigned short* wqr_bf   = take((size_t)D * U);
    unsigned short* wkr_bf   = take((size_t)Z * 64);
    unsigned short* wfc_bf   = take((size_t)Z * Z);
    unsigned short* ckv_bf   = take((size_t)MS * D);
    unsigned short* cq_bf    = take((size_t)MS * D);
    unsigned short* ktc_bf   = take((size_t)MS * U);
    unsigned short* vtc_bf   = take((size_t)MS * U);
    unsigned short* qtc_bf   = take((size_t)MS * U);
    unsigned short* qr_bf    = take((size_t)MS * U);
    unsigned short* kr_bf    = take((size_t)MS * 64);
    unsigned short* attn_bf  = take((size_t)MS * Z);

    auto cvt = [&](const float* src, unsigned short* dst, size_t n) {
        int blocks = (int)((n / 4 + 255) / 256);
        cvt_f32_bf16_k<<<blocks, 256, 0, stream>>>(src, dst, (int)n);
    };
    cvt(key,   key_bf,   (size_t)MS * Z);
    cvt(query, query_bf, (size_t)MS * Z);
    cvt(w_dkv, wdkv_bf, (size_t)Z * D);
    cvt(w_uk,  wuk_bf,  (size_t)D * U);
    cvt(w_uv,  wuv_bf,  (size_t)D * U);
    cvt(w_dq,  wdq_bf,  (size_t)Z * D);
    cvt(w_uq,  wuq_bf,  (size_t)D * U);
    cvt(w_qr,  wqr_bf,  (size_t)D * U);
    cvt(w_kr,  wkr_bf,  (size_t)Z * 64);
    cvt(w_fc,  wfc_bf,  (size_t)Z * Z);

    auto gemm_bf = [&](const unsigned short* A, const unsigned short* Bm,
                       const float* bias, unsigned short* C, int M, int N, int K) {
        dim3 g((N + 127) / 128, M / 128);
        gemm_bf16_k<unsigned short><<<g, 256, 0, stream>>>(A, Bm, bias, C, M, N, K);
    };

    // projections
    gemm_bf(key_bf,   wdkv_bf, b_dkv, ckv_bf, MS, D, Z);   // c_t_kv
    gemm_bf(ckv_bf,   wuk_bf,  b_uk,  ktc_bf, MS, U, D);   // k_t_c
    gemm_bf(ckv_bf,   wuv_bf,  b_uv,  vtc_bf, MS, U, D);   // v_t_c
    gemm_bf(query_bf, wdq_bf,  b_dq,  cq_bf,  MS, D, Z);   // c_t_q
    gemm_bf(cq_bf,    wuq_bf,  b_uq,  qtc_bf, MS, U, D);   // q_t_c
    gemm_bf(cq_bf,    wqr_bf,  b_qr,  qr_bf,  MS, U, D);   // q_t_r (pre-rope)
    gemm_bf(key_bf,   wkr_bf,  b_kr,  kr_bf,  MS, 64, Z);  // k_t_r (pre-rope)

    // rope (in place)
    {
        int pq = B * S * H * 32;
        rope_inplace_k<<<(pq + 255) / 256, 256, 0, stream>>>(qr_bf, S, H, pq);
        int pk = B * S * 1 * 32;
        rope_inplace_k<<<(pk + 255) / 256, 256, 0, stream>>>(kr_bf, S, 1, pk);
    }

    // flash attention (64-key tiles)
    {
        dim3 g(S / 128, H, B);
        mla_attn_k<<<g, 256, 0, stream>>>(qtc_bf, qr_bf, ktc_bf, kr_bf, vtc_bf,
                                          attn_bf, S);
    }

    // output projection -> f32 d_out
    {
        dim3 g((Z + 127) / 128, MS / 128);
        gemm_bf16_k<float><<<g, 256, 0, stream>>>(attn_bf, wfc_bf, b_fc, outp,
                                                  MS, Z, Z);
    }
}
